// SelfAttentionModule_4733053960732
// MI455X (gfx1250) — matrix operand order repaired and checked by reading, not compiled, not run on hardware
//
#include <hip/hip_runtime.h>
#include <hip/hip_bf16.h>
#include <cstdint>
#include <cstddef>

// ---------------------------------------------------------------------------
// Problem constants
// ---------------------------------------------------------------------------
static constexpr int BB  = 64;    // batch
static constexpr int CIN = 256;   // input channels
static constexpr int CP  = 128;   // projected channels
static constexpr int NN  = 1024;  // H*W
static constexpr float EPSV = 1e-5f;

typedef __bf16 bf16;
typedef bf16  v8bf  __attribute__((ext_vector_type(8)));
typedef bf16  v16bf __attribute__((ext_vector_type(16)));
typedef float v8f   __attribute__((ext_vector_type(8)));

#define DEVFN __device__ __forceinline__

// ---------------------------------------------------------------------------
// Workspace layout (bytes). All big buffers 256B aligned by construction.
// ---------------------------------------------------------------------------
static constexpr size_t OFF_XBF   = 0;                                    // [B][N][CIN] bf16
static constexpr size_t OFF_W3    = OFF_XBF  + (size_t)BB*NN*CIN*2;       // [384][256] bf16
static constexpr size_t OFF_WO    = OFF_W3   + (size_t)384*256*2;         // [256][128] bf16
static constexpr size_t OFF_GS3   = OFF_WO   + (size_t)256*128*2;         // [384] f32 BN gamma
static constexpr size_t OFF_GB3   = OFF_GS3  + 384*4;                     // [384] f32 BN beta
static constexpr size_t OFF_GSO   = OFF_GB3  + 384*4;                     // [256]
static constexpr size_t OFF_GBO   = OFF_GSO  + 256*4;                     // [256]
static constexpr size_t OFF_MEAN3 = OFF_GBO  + 256*4;                     // [384]
static constexpr size_t OFF_RSTD3 = OFF_MEAN3+ 384*4;                     // [384]
static constexpr size_t OFF_MEANO = OFF_RSTD3+ 384*4;                     // [256]
static constexpr size_t OFF_RSTDO = OFF_MEANO+ 256*4;                     // [256]
static constexpr size_t OFF_MKEY  = OFF_RSTDO+ 256*4;                     // [64] u32 ordered-max keys
static constexpr size_t OFF_SUME  = OFF_MKEY + 64*4;                      // [64] f32 sum(exp)
static constexpr size_t OFF_PART  = OFF_SUME + 64*4;                      // [64*256] partial sums
static constexpr size_t OFF_RAW3  = OFF_PART + (size_t)64*256*4;          // [B][384][N] f32
static constexpr size_t OFF_ORAW  = OFF_RAW3;                             // reuse (dead by then) [B][256][N] f32
static constexpr size_t OFF_TBF   = OFF_RAW3 + (size_t)BB*384*NN*4;       // [B][N][CP] bf16
static constexpr size_t OFF_PBF   = OFF_TBF  + (size_t)BB*NN*CP*2;        // [B][N][CP] bf16
static constexpr size_t OFF_GBF   = OFF_PBF  + (size_t)BB*NN*CP*2;        // [B][CP][N] bf16
static constexpr size_t OFF_ATTN  = OFF_GBF  + (size_t)BB*CP*NN*2;        // [B][N][N] bf16
static constexpr size_t OFF_SAT   = OFF_ATTN + (size_t)BB*NN*NN*2;        // [B][N][CP] bf16

// ---------------------------------------------------------------------------
// Ordered-float <-> uint keys for atomicMax over floats (incl. negatives)
// ---------------------------------------------------------------------------
DEVFN unsigned fkey(float x) {
  unsigned u = __float_as_uint(x);
  return (u & 0x80000000u) ? ~u : (u | 0x80000000u);
}
DEVFN float fdecode(unsigned k) {
  return (k & 0x80000000u) ? __uint_as_float(k ^ 0x80000000u)
                           : __uint_as_float(~k);
}

// ---------------------------------------------------------------------------
// Per-wave 16x16 row-tile x NTT col-tiles GEMM core (bf16 -> f32 WMMA).
// A: row-major [16 x K], lda elements; lane m = lane%16, contiguous k order (same as B):
//    VGPRs 0..7 hold K = half*16..half*16+15 (two 16B loads).
// B: K-contiguous per column (column n at Bm + n*ldb); lane n = lane%16,
//    VGPRs hold K = half*16..half*16+15 (one 32B load).
// ---------------------------------------------------------------------------
union ABu { v16bf v; v8bf h[2]; };

template <int NTT>
DEVFN void wmma_rowtile(const bf16* __restrict__ A, int lda,
                        const bf16* __restrict__ Bm, int ldb,
                        int K, v8f acc[NTT]) {
  const int lane = threadIdx.x & 31;
  const int col  = lane & 15;
  const int hi   = lane >> 4;
  const bf16* a_base = A  + (size_t)col * lda + hi * 16;
  const bf16* b_base = Bm + (size_t)col * ldb + hi * 16;
  for (int k = 0; k < K; k += 32) {
    ABu a;
    a.h[0] = *(const v8bf*)(a_base + k);
    a.h[1] = *(const v8bf*)(a_base + k + 8);
#pragma unroll
    for (int t = 0; t < NTT; ++t) {
      v16bf b = *(const v16bf*)(b_base + (size_t)t * 16 * ldb + k);
      acc[t] = __builtin_amdgcn_wmma_f32_16x16x32_bf16(
          false, a.v, false, b, (short)0, acc[t], false, false);
    }
  }
}

// ---------------------------------------------------------------------------
// Small setup kernels
// ---------------------------------------------------------------------------
__global__ void k_init(unsigned* mkey, float* sume) {
  int i = threadIdx.x;
  if (i < BB) { mkey[i] = 0u; sume[i] = 0.0f; }
}

__global__ void k_pack(const float* __restrict__ tw, const float* __restrict__ pw,
                       const float* __restrict__ gw, const float* __restrict__ ow,
                       const float* __restrict__ tg, const float* __restrict__ tb,
                       const float* __restrict__ pg, const float* __restrict__ pb,
                       const float* __restrict__ gg, const float* __restrict__ gb,
                       const float* __restrict__ og, const float* __restrict__ ob,
                       bf16* __restrict__ w3, bf16* __restrict__ wo,
                       float* __restrict__ gs3, float* __restrict__ gb3,
                       float* __restrict__ gso, float* __restrict__ gbo) {
  int tid = blockIdx.x * blockDim.x + threadIdx.x;
  if (tid < 384 * 256) {
    int o = tid >> 8, c = tid & 255;
    float v = (o < 128) ? tw[o * 256 + c]
            : (o < 256) ? pw[(o - 128) * 256 + c]
                        : gw[(o - 256) * 256 + c];
    w3[tid] = (bf16)v;
  } else if (tid < 384 * 256 + 256 * 128) {
    int i = tid - 384 * 256;
    wo[i] = (bf16)ow[i];
  } else if (tid < 384 * 256 + 256 * 128 + 384) {
    int i = tid - (384 * 256 + 256 * 128);
    gs3[i] = (i < 128) ? tg[i] : (i < 256) ? pg[i - 128] : gg[i - 256];
    gb3[i] = (i < 128) ? tb[i] : (i < 256) ? pb[i - 128] : gb[i - 256];
  } else if (tid < 384 * 256 + 256 * 128 + 384 + 256) {
    int i = tid - (384 * 256 + 256 * 128 + 384);
    gso[i] = og[i];
    gbo[i] = ob[i];
  }
}

// x [B][CIN][N] f32 -> xbf [B][N][CIN] bf16 (K-contiguous for the proj GEMM)
__global__ void k_cast_x(const float* __restrict__ x, bf16* __restrict__ xbf) {
  int tid = blockIdx.x * blockDim.x + threadIdx.x;  // B*NN*CIN = 2^24 threads
  int c = tid & 255;
  int n = (tid >> 8) & 1023;
  int b = tid >> 18;
  xbf[tid] = (bf16)x[((size_t)b * CIN + c) * NN + n];
}

// ---------------------------------------------------------------------------
// GEMM 1: raw[b][o][n] = sum_c w3[o][c] * xbf[b][n][c]   (o in 0..383, K=256)
// ---------------------------------------------------------------------------
__global__ void k_gemm_proj(const bf16* __restrict__ w3,
                            const bf16* __restrict__ xbf,
                            float* __restrict__ raw) {
  int wave = (blockIdx.x * blockDim.x + threadIdx.x) >> 5;
  int tgrp = wave & 15;            // 16 col groups * 4 tiles = 64 n-tiles
  int tmp  = wave >> 4;
  int tm   = tmp % 24;             // 24 row tiles (384/16)
  int b    = tmp / 24;
  const bf16* A  = w3  + (size_t)tm * 16 * CIN;
  const bf16* Bm = xbf + ((size_t)b * NN + (size_t)tgrp * 64) * CIN;
  v8f acc[4] = {};
  wmma_rowtile<4>(A, CIN, Bm, CIN, CIN, acc);
  int lane = threadIdx.x & 31, col = lane & 15, hi = lane >> 4;
#pragma unroll
  for (int t = 0; t < 4; ++t)
#pragma unroll
    for (int r = 0; r < 8; ++r) {
      int o = tm * 16 + r + 8 * hi;
      int n = tgrp * 64 + t * 16 + col;
      raw[((size_t)b * 384 + o) * NN + n] = acc[t][r];
    }
}

// ---------------------------------------------------------------------------
// Per-channel BN stats over (B, N): one block per channel (biased variance)
// ---------------------------------------------------------------------------
__global__ void k_stats(const float* __restrict__ raw, int C,
                        float* __restrict__ mean, float* __restrict__ rstd) {
  int c = blockIdx.x;
  const int total = BB * NN;
  float s = 0.0f, ss = 0.0f;
  for (int i = threadIdx.x; i < total; i += blockDim.x) {
    int b = i >> 10, n = i & 1023;
    float v = raw[((size_t)b * C + c) * NN + n];
    s += v; ss += v * v;
  }
  __shared__ float sh[256], sh2[256];
  sh[threadIdx.x] = s; sh2[threadIdx.x] = ss;
  __syncthreads();
  for (int o = 128; o > 0; o >>= 1) {
    if ((int)threadIdx.x < o) { sh[threadIdx.x] += sh[threadIdx.x + o]; sh2[threadIdx.x] += sh2[threadIdx.x + o]; }
    __syncthreads();
  }
  if (threadIdx.x == 0) {
    float m = sh[0] / (float)total;
    float v = sh2[0] / (float)total - m * m;
    mean[c] = m;
    rstd[c] = rsqrtf(fmaxf(v, 0.0f) + EPSV);
  }
}

// ---------------------------------------------------------------------------
// Normalize + cast: raw[b][o][n] -> t/p as [B][N][CP], g as [B][CP][N]
// ---------------------------------------------------------------------------
__global__ void k_norm_tpg(const float* __restrict__ raw,
                           const float* __restrict__ mean, const float* __restrict__ rstd,
                           const float* __restrict__ gs, const float* __restrict__ gb,
                           bf16* __restrict__ tbf, bf16* __restrict__ pbf,
                           bf16* __restrict__ gbf) {
  size_t tid = (size_t)blockIdx.x * blockDim.x + threadIdx.x;  // B*384*NN
  int n = (int)(tid & 1023);
  int o = (int)((tid >> 10) % 384);
  int b = (int)(tid / ((size_t)384 * 1024));
  float v = raw[tid];
  float y = (v - mean[o]) * rstd[o] * gs[o] + gb[o];
  int proj = o >> 7, cp = o & 127;
  if (proj == 0)      tbf[((size_t)b * NN + n) * CP + cp] = (bf16)y;
  else if (proj == 1) pbf[((size_t)b * NN + n) * CP + cp] = (bf16)y;
  else                gbf[((size_t)b * CP + cp) * NN + n] = (bf16)y;
}

// ---------------------------------------------------------------------------
// GEMM 2: attn[b][n][m] = sum_cp t[b][n][cp] * p[b][m][cp]   (K=128)
// Also tracks the per-batch max via wave shuffle + ordered atomicMax.
// ---------------------------------------------------------------------------
__global__ void k_gemm_attn(const bf16* __restrict__ tbf,
                            const bf16* __restrict__ pbf,
                            bf16* __restrict__ attn,
                            unsigned* __restrict__ mkey) {
  int wave = (blockIdx.x * blockDim.x + threadIdx.x) >> 5;
  int tgrp = wave & 15;
  int tmp  = wave >> 4;
  int tm   = tmp & 63;
  int b    = tmp >> 6;
  const bf16* A  = tbf + ((size_t)b * NN + (size_t)tm * 16) * CP;
  const bf16* Bm = pbf + ((size_t)b * NN + (size_t)tgrp * 64) * CP;
  v8f acc[4] = {};
  wmma_rowtile<4>(A, CP, Bm, CP, CP, acc);
  int lane = threadIdx.x & 31, col = lane & 15, hi = lane >> 4;
  float lmax = -3.0e38f;
#pragma unroll
  for (int t = 0; t < 4; ++t)
#pragma unroll
    for (int r = 0; r < 8; ++r) {
      float v = acc[t][r];
      lmax = fmaxf(lmax, v);
      int n = tm * 16 + r + 8 * hi;
      int m = tgrp * 64 + t * 16 + col;
      attn[((size_t)b * NN + n) * NN + m] = (bf16)v;
    }
#pragma unroll
  for (int off = 16; off > 0; off >>= 1)
    lmax = fmaxf(lmax, __shfl_xor(lmax, off, 32));
  if (lane == 0) atomicMax(mkey + b, fkey(lmax));
}

// ---------------------------------------------------------------------------
// exp(attn - max) in place (bf16), deterministic two-level sum
// ---------------------------------------------------------------------------
__global__ void k_expsum(bf16* __restrict__ attn,
                         const unsigned* __restrict__ mkey,
                         float* __restrict__ part) {
  int b = blockIdx.x >> 8;          // 256 chunks per batch, 4096 elems each
  int chunk = blockIdx.x & 255;
  float vmax = fdecode(mkey[b]);
  bf16* base = attn + ((size_t)b << 20) + ((size_t)chunk << 12);
  float s = 0.0f;
  for (int i = threadIdx.x; i < 4096; i += 256) {
    float e = __expf((float)base[i] - vmax);
    base[i] = (bf16)e;
    s += e;
  }
  __shared__ float sh[256];
  sh[threadIdx.x] = s;
  __syncthreads();
  for (int o = 128; o > 0; o >>= 1) {
    if ((int)threadIdx.x < o) sh[threadIdx.x] += sh[threadIdx.x + o];
    __syncthreads();
  }
  if (threadIdx.x == 0) part[blockIdx.x] = sh[0];
}

__global__ void k_sumreduce(const float* __restrict__ part, float* __restrict__ sume) {
  __shared__ float sh[256];
  sh[threadIdx.x] = part[(size_t)blockIdx.x * 256 + threadIdx.x];
  __syncthreads();
  for (int o = 128; o > 0; o >>= 1) {
    if ((int)threadIdx.x < o) sh[threadIdx.x] += sh[threadIdx.x + o];
    __syncthreads();
  }
  if (threadIdx.x == 0) sume[blockIdx.x] = sh[0];
}

// ---------------------------------------------------------------------------
// GEMM 3: sat[b][n][cp] = (1/sum_b) * sum_m E[b][n][m] * g[b][cp][m]  (K=1024)
// NT=8 so each wave covers all 128 cp columns: attn (134 MB) is read once.
// ---------------------------------------------------------------------------
__global__ void k_gemm_sat(const bf16* __restrict__ attn,
                           const bf16* __restrict__ gbf,
                           const float* __restrict__ sume,
                           bf16* __restrict__ satb) {
  int wave = (blockIdx.x * blockDim.x + threadIdx.x) >> 5;
  int tm = wave & 63;
  int b  = wave >> 6;
  const bf16* A  = attn + ((size_t)b * NN + (size_t)tm * 16) * NN;
  const bf16* Bm = gbf + (size_t)b * CP * NN;
  v8f acc[8] = {};
  wmma_rowtile<8>(A, NN, Bm, NN, NN, acc);
  float inv = 1.0f / sume[b];
  int lane = threadIdx.x & 31, col = lane & 15, hi = lane >> 4;
#pragma unroll
  for (int t = 0; t < 8; ++t)
#pragma unroll
    for (int r = 0; r < 8; ++r) {
      int n  = tm * 16 + r + 8 * hi;
      int cp = t * 16 + col;
      satb[((size_t)b * NN + n) * CP + cp] = (bf16)(acc[t][r] * inv);
    }
}

// ---------------------------------------------------------------------------
// GEMM 4: oraw[b][ci][n] = sum_cp wo[ci][cp] * sat[b][n][cp]   (K=128)
// ---------------------------------------------------------------------------
__global__ void k_gemm_omega(const bf16* __restrict__ wo,
                             const bf16* __restrict__ satb,
                             float* __restrict__ oraw) {
  int wave = (blockIdx.x * blockDim.x + threadIdx.x) >> 5;
  int tgrp = wave & 15;
  int tmp  = wave >> 4;
  int tm   = tmp & 15;              // 16 row tiles (256/16)
  int b    = tmp >> 4;
  const bf16* A  = wo   + (size_t)tm * 16 * CP;
  const bf16* Bm = satb + ((size_t)b * NN + (size_t)tgrp * 64) * CP;
  v8f acc[4] = {};
  wmma_rowtile<4>(A, CP, Bm, CP, CP, acc);
  int lane = threadIdx.x & 31, col = lane & 15, hi = lane >> 4;
#pragma unroll
  for (int t = 0; t < 4; ++t)
#pragma unroll
    for (int r = 0; r < 8; ++r) {
      int ci = tm * 16 + r + 8 * hi;
      int n  = tgrp * 64 + t * 16 + col;
      oraw[((size_t)b * CIN + ci) * NN + n] = acc[t][r];
    }
}

// ---------------------------------------------------------------------------
// Final: out = x + BN(oraw) (fp32)
// ---------------------------------------------------------------------------
__global__ void k_final(const float* __restrict__ x, const float* __restrict__ oraw,
                        const float* __restrict__ mean, const float* __restrict__ rstd,
                        const float* __restrict__ og, const float* __restrict__ ob,
                        float* __restrict__ out) {
  size_t tid = (size_t)blockIdx.x * blockDim.x + threadIdx.x;  // B*CIN*NN
  int ci = (int)((tid >> 10) & 255);
  float v = oraw[tid];
  out[tid] = x[tid] + (v - mean[ci]) * rstd[ci] * og[ci] + ob[ci];
}

// ---------------------------------------------------------------------------
// Host launcher
// ---------------------------------------------------------------------------
extern "C" void kernel_launch(void* const* d_in, const int* in_sizes, int n_in,
                              void* d_out, int out_size, void* d_ws, size_t ws_size,
                              hipStream_t stream) {
  (void)in_sizes; (void)n_in; (void)out_size; (void)ws_size;

  const float* x        = (const float*)d_in[0];
  const float* theta_w  = (const float*)d_in[1];
  const float* theta_g  = (const float*)d_in[3];
  const float* theta_be = (const float*)d_in[4];
  const float* phi_w    = (const float*)d_in[5];
  const float* phi_g    = (const float*)d_in[7];
  const float* phi_be   = (const float*)d_in[8];
  const float* gamma_w  = (const float*)d_in[9];
  const float* gamma_g  = (const float*)d_in[11];
  const float* gamma_be = (const float*)d_in[12];
  const float* omega_w  = (const float*)d_in[13];
  const float* omega_g  = (const float*)d_in[15];
  const float* omega_be = (const float*)d_in[16];
  // conv biases (d_in[2],[6],[10],[14]) are absorbed by BN mean subtraction.

  char* ws = (char*)d_ws;
  bf16*     xbf   = (bf16*)(ws + OFF_XBF);
  bf16*     w3    = (bf16*)(ws + OFF_W3);
  bf16*     wo    = (bf16*)(ws + OFF_WO);
  float*    gs3   = (float*)(ws + OFF_GS3);
  float*    gb3   = (float*)(ws + OFF_GB3);
  float*    gso   = (float*)(ws + OFF_GSO);
  float*    gbo   = (float*)(ws + OFF_GBO);
  float*    mean3 = (float*)(ws + OFF_MEAN3);
  float*    rstd3 = (float*)(ws + OFF_RSTD3);
  float*    meano = (float*)(ws + OFF_MEANO);
  float*    rstdo = (float*)(ws + OFF_RSTDO);
  unsigned* mkey  = (unsigned*)(ws + OFF_MKEY);
  float*    sume  = (float*)(ws + OFF_SUME);
  float*    part  = (float*)(ws + OFF_PART);
  float*    raw3  = (float*)(ws + OFF_RAW3);
  float*    oraw  = (float*)(ws + OFF_ORAW);
  bf16*     tbf   = (bf16*)(ws + OFF_TBF);
  bf16*     pbf   = (bf16*)(ws + OFF_PBF);
  bf16*     gbf   = (bf16*)(ws + OFF_GBF);
  bf16*     attn  = (bf16*)(ws + OFF_ATTN);
  bf16*     satb  = (bf16*)(ws + OFF_SAT);

  // 0) init per-batch softmax scalars
  k_init<<<1, 64, 0, stream>>>(mkey, sume);

  // 1) pack weights + BN affine params
  k_pack<<<515, 256, 0, stream>>>(theta_w, phi_w, gamma_w, omega_w,
                                  theta_g, theta_be, phi_g, phi_be,
                                  gamma_g, gamma_be, omega_g, omega_be,
                                  w3, wo, gs3, gb3, gso, gbo);

  // 2) x -> bf16 [B][N][CIN]
  k_cast_x<<<65536, 256, 0, stream>>>(x, xbf);

  // 3) projection GEMM (theta|phi|gamma fused): 64*24*16 waves
  k_gemm_proj<<<3072, 256, 0, stream>>>(w3, xbf, raw3);

  // 4) BN stats over (B,N) per channel of the 384 projected channels
  k_stats<<<384, 256, 0, stream>>>(raw3, 384, mean3, rstd3);

  // 5) normalize + cast + layout split
  k_norm_tpg<<<98304, 256, 0, stream>>>(raw3, mean3, rstd3, gs3, gb3, tbf, pbf, gbf);

  // 6) attention scores + global max: 64*64*16 waves
  k_gemm_attn<<<8192, 256, 0, stream>>>(tbf, pbf, attn, mkey);

  // 7) exp in place + deterministic sum
  k_expsum<<<BB * 256, 256, 0, stream>>>(attn, mkey, part);
  k_sumreduce<<<BB, 256, 0, stream>>>(part, sume);

  // 8) sat = softmax(attn) @ g^T : 64*64 waves (NT=8 covers all 128 cols)
  k_gemm_sat<<<512, 256, 0, stream>>>(attn, gbf, sume, satb);

  // 9) omega projection back to CIN: 64*16*16 waves
  k_gemm_omega<<<2048, 256, 0, stream>>>(wo, satb, oraw);

  // 10) BN stats on omega output
  k_stats<<<256, 256, 0, stream>>>(oraw, 256, meano, rstdo);

  // 11) BN + residual -> fp32 output
  k_final<<<65536, 256, 0, stream>>>(x, oraw, meano, rstdo, gso, gbo, (float*)d_out);
}
